// DeepSeekMoE_21294447853771
// MI455X (gfx1250) — matrix operand order repaired and checked by reading, not compiled, not run on hardware
//
#include <hip/hip_runtime.h>
#include <hip/hip_bf16.h>

// Problem constants (from reference): B=8, S=2048, H=1024, I=2048, E=8, TOP_K=2
#define N_TOK  (8 * 2048)
#define HDIM   1024
#define IDIM   2048
#define NEXP   8

typedef __attribute__((ext_vector_type(16))) __bf16 v16bf;
typedef __attribute__((ext_vector_type(8)))  __bf16 bf8;
typedef __attribute__((ext_vector_type(4)))  __bf16 bf4;
typedef __attribute__((ext_vector_type(8)))  float  v8f;
typedef __attribute__((ext_vector_type(4)))  unsigned int v4u;
typedef __attribute__((ext_vector_type(8)))  int v8i;
typedef __attribute__((ext_vector_type(4)))  int v4i;

union FragU { v16bf v; bf8 h[2]; };

__device__ inline v8f vzero() {
  v8f z;
#pragma unroll
  for (int i = 0; i < 8; ++i) z[i] = 0.0f;
  return z;
}

// A-matrix 16x32 bf16 fragment from a row-major [16][pitch] tile in LDS.
// CDNA5 layout: lane<16  -> M=lane,    K runs [k0..k0+7]   and [k0+16..k0+23]
//               lane>=16 -> M=lane-16, K runs [k0+8..k0+15] and [k0+24..k0+31]
__device__ inline v16bf load_a_frag(const __bf16* base, int pitch, int k0, int lane) {
  int m  = lane & 15;
  int ks = (lane >> 4) * 8;
  const __bf16* p = base + m * pitch + k0 + ks;
  FragU f;
  f.h[0] = *(const bf8*)(p);
  f.h[1] = *(const bf8*)(p + 16);
  return f.v;
}

// B-matrix 32x16 bf16 fragment from row-major weights W[out][in] (rows contiguous in K).
// CDNA5 layout: lane<16 -> N=lane, K=[k0..k0+7],[k0+16..k0+23]; lane>=16 -> N=lane-16, K=[k0+8..k0+15],[k0+24..k0+31]
__device__ inline v16bf load_b_frag(const __bf16* w, int ldk, int n0, int k0, int lane) {
  int n  = lane & 15;
  int ks = (lane >> 4) * 8;
  const __bf16* p = w + (size_t)(n0 + n) * ldk + k0 + ks;
  FragU f;
  f.h[0] = *(const bf8*)(p);
  f.h[1] = *(const bf8*)(p + 16);
  return f.v;
}

// ---------------------------------------------------------------------------
// TDM: async DMA of a [rows x HDIM] bf16 tile (row stride HDIM) into LDS.
// Hand-built D# per CDNA5 ISA ch.8 (groups 0/1; groups 2/3 zero for 2-D).
// Tracked by TENSORcnt; waited with s_wait_tensorcnt.
// Toolchain here is the 6-arg (clang-23 / therock) builtin flavor.
// ---------------------------------------------------------------------------
__device__ inline void tdm_load_tile(const __bf16* gsrc, unsigned lds_off,
                                     unsigned rows, unsigned tensor_rows) {
  unsigned long long ga = (unsigned long long)(uintptr_t)gsrc;
  v4u g0;
  g0.x = 1u;                                              // count=1, user D#
  g0.y = lds_off;                                         // lds_addr (bytes)
  g0.z = (unsigned)(ga & 0xffffffffu);                    // global_addr[31:0]
  g0.w = (unsigned)((ga >> 32) & 0x01ffffffu) | (2u << 30); // addr[56:32] | type=2
  v8i g1;
  g1[0] = (int)(1u << 16);                                // data_size=1 (2 bytes)
  g1[1] = (int)((HDIM & 0xffffu) << 16);                  // tensor_dim0 lo16 @ [63:48]
  g1[2] = (int)((HDIM >> 16) & 0xffffu) |
          (int)((tensor_rows & 0xffffu) << 16);           // dim0 hi | tensor_dim1 lo
  g1[3] = (int)((tensor_rows >> 16) & 0xffffu) |
          (int)((HDIM & 0xffffu) << 16);                  // tensor_dim1 hi | tile_dim0
  g1[4] = (int)(rows & 0xffffu);                          // tile_dim1 (tile_dim2=0)
  g1[5] = (int)HDIM;                                      // tensor_dim0_stride lo32
  g1[6] = (int)((HDIM & 0xffffu) << 16);                  // stride0 hi=0 | stride1 lo16
  g1[7] = 0;                                              // stride1 hi
  v4i z4 = {0, 0, 0, 0};
  v8i z8 = {0, 0, 0, 0, 0, 0, 0, 0};
  __builtin_amdgcn_tensor_load_to_lds(g0, g1, z4, z4, z8, 0);
}

// ---------------------------------------------------------------------------
// fp32 -> bf16 conversion (grid-stride, 4 elems / thread / iter)
// ---------------------------------------------------------------------------
__global__ void cvt_kernel(const float* __restrict__ src, __bf16* __restrict__ dst,
                           long long n) {
  long long stride = (long long)gridDim.x * blockDim.x * 4;
  for (long long j = ((long long)blockIdx.x * blockDim.x + threadIdx.x) * 4; j < n;
       j += stride) {
    float4 f = *(const float4*)(src + j);
    bf4 o = {(__bf16)f.x, (__bf16)f.y, (__bf16)f.z, (__bf16)f.w};
    *(bf4*)(dst + j) = o;
  }
}

// ---------------------------------------------------------------------------
// Router: sigmoid top-2 -> dense combine weights cw[N_TOK][NEXP]
// ---------------------------------------------------------------------------
__global__ __launch_bounds__(256) void router_kernel(const float* __restrict__ x,
                                                     const float* __restrict__ Wr,
                                                     const float* __restrict__ rb,
                                                     float* __restrict__ cw) {
  int token = blockIdx.x * 8 + (threadIdx.x >> 5);
  int lane  = threadIdx.x & 31;
  float acc[NEXP];
#pragma unroll
  for (int e = 0; e < NEXP; ++e) acc[e] = 0.0f;
  const float* xr = x + (size_t)token * HDIM;
  for (int i = lane; i < HDIM; i += 32) {
    float xv = xr[i];
#pragma unroll
    for (int e = 0; e < NEXP; ++e) acc[e] += xv * Wr[e * HDIM + i];
  }
#pragma unroll
  for (int e = 0; e < NEXP; ++e) {
#pragma unroll
    for (int off = 16; off > 0; off >>= 1) acc[e] += __shfl_xor(acc[e], off, 32);
  }
  if (lane == 0) {
    float p[NEXP];
#pragma unroll
    for (int e = 0; e < NEXP; ++e) {
      float l = acc[e] + rb[e];
      p[e] = __builtin_amdgcn_rcpf(1.0f + __expf(-l));
    }
    int i1 = 0; float p1 = p[0];
#pragma unroll
    for (int e = 1; e < NEXP; ++e)
      if (p[e] > p1) { p1 = p[e]; i1 = e; }
    int i2 = -1; float p2 = -1.0f;
#pragma unroll
    for (int e = 0; e < NEXP; ++e)
      if (e != i1 && p[e] > p2) { p2 = p[e]; i2 = e; }
#pragma unroll
    for (int e = 0; e < NEXP; ++e)
      cw[(size_t)token * NEXP + e] = (e == i1) ? p1 : ((e == i2) ? p2 : 0.0f);
  }
}

// ---------------------------------------------------------------------------
// Fused MoE kernel. Block = 32-token tile, 256 threads = 8 waves.
// x tile arrives via TDM; hidden tile (32 x 2048 bf16) lives in LDS; output
// tile (32 x 1024 f32) accumulates in registers across all 9 experts.
// Each B fragment feeds two WMMAs (two 16-row A tiles) -> halves L2 traffic.
// ---------------------------------------------------------------------------
#define MTILE  32
#define XPITCH 1024                       // TDM packs rows contiguously
#define HPITCH 2056                       // padded, 16B-aligned rows
#define MOE_SMEM ((MTILE * XPITCH + MTILE * HPITCH) * 2 + 128)

__global__ __launch_bounds__(256, 1) void moe_kernel(
    const __bf16* __restrict__ xb,
    const __bf16* __restrict__ Wg_s, const __bf16* __restrict__ Wu_s,
    const __bf16* __restrict__ Wd_s,
    const __bf16* __restrict__ Wg, const __bf16* __restrict__ Wu,
    const __bf16* __restrict__ Wd,
    const float* __restrict__ cw, float* __restrict__ out) {
  extern __shared__ char smem[];
  __bf16* xs  = (__bf16*)smem;                   // [32][1024] bf16 x tile
  __bf16* hs  = xs + MTILE * XPITCH;             // [32][HPITCH] bf16 hidden tile
  float*  cws = (float*)(hs + MTILE * HPITCH);   // [32] combine weights

  const int token0 = blockIdx.x * MTILE;
  const int wave   = threadIdx.x >> 5;
  const int lane   = threadIdx.x & 31;

  // Async DMA the 32 x 1024 bf16 x tile into LDS (one TDM op from wave 0).
  if (wave == 0) {
    tdm_load_tile(xb + (size_t)token0 * HDIM, (unsigned)(uintptr_t)xs,
                  MTILE, N_TOK);
  }

  // Persistent f32 accumulators: wave owns 128 of 1024 out channels x 32 rows.
  v8f oacc0[8], oacc1[8];
#pragma unroll
  for (int t = 0; t < 8; ++t) { oacc0[t] = vzero(); oacc1[t] = vzero(); }

  if (wave == 0) __builtin_amdgcn_s_wait_tensorcnt(0);

  for (int s = 0; s < 1 + NEXP; ++s) {
    __syncthreads();  // x tile ready (iter 0); hs/cws drained (iters > 0)
    if (threadIdx.x < MTILE)
      cws[threadIdx.x] =
          (s == 0) ? 1.0f : cw[(size_t)(token0 + threadIdx.x) * NEXP + (s - 1)];
    __syncthreads();

    // Uniform skip of experts this tile never routed to.
    float mx = 0.0f;
#pragma unroll
    for (int m = 0; m < MTILE; ++m) mx = fmaxf(mx, cws[m]);
    if (s > 0 && mx == 0.0f) continue;

    const __bf16* wg = (s == 0) ? Wg_s : Wg + (size_t)(s - 1) * IDIM * HDIM;
    const __bf16* wu = (s == 0) ? Wu_s : Wu + (size_t)(s - 1) * IDIM * HDIM;
    const __bf16* wd = (s == 0) ? Wd_s : Wd + (size_t)(s - 1) * HDIM * IDIM;

    // Warm L2 with next expert's weights while this one computes.
    if (s <= NEXP - 1) {
      __builtin_prefetch(Wg + (size_t)s * IDIM * HDIM, 0, 1);
      __builtin_prefetch(Wu + (size_t)s * IDIM * HDIM, 0, 1);
      __builtin_prefetch(Wd + (size_t)s * HDIM * IDIM, 0, 1);
    }

    // ---- Stage 1: hidden = cw * silu(x@Wg^T) * (x@Wu^T) --------------------
    // Wave computes its 32x256 slice; each B frag feeds both 16-row A tiles.
    for (int nt = 0; nt < 16; ++nt) {
      int n0 = wave * 256 + nt * 16;
      v8f ag0 = vzero(), au0 = vzero(), ag1 = vzero(), au1 = vzero();
      for (int k0 = 0; k0 < HDIM; k0 += 32) {
        v16bf a0 = load_a_frag(xs, XPITCH, k0, lane);
        v16bf a1 = load_a_frag(xs + 16 * XPITCH, XPITCH, k0, lane);
        v16bf bg = load_b_frag(wg, HDIM, n0, k0, lane);
        v16bf bu = load_b_frag(wu, HDIM, n0, k0, lane);
        ag0 = __builtin_amdgcn_wmma_f32_16x16x32_bf16(false, a0, false, bg,
                                                      (short)0, ag0, false, false);
        au0 = __builtin_amdgcn_wmma_f32_16x16x32_bf16(false, a0, false, bu,
                                                      (short)0, au0, false, false);
        ag1 = __builtin_amdgcn_wmma_f32_16x16x32_bf16(false, a1, false, bg,
                                                      (short)0, ag1, false, false);
        au1 = __builtin_amdgcn_wmma_f32_16x16x32_bf16(false, a1, false, bu,
                                                      (short)0, au1, false, false);
      }
      // C/D layout: element r -> M = (lane>=16 ? 8 : 0) + r, N = lane&15
      int n = lane & 15, mbase = (lane >> 4) * 8;
#pragma unroll
      for (int r = 0; r < 8; ++r) {
        float g = ag0[r], u = au0[r];
        float h = g * __builtin_amdgcn_rcpf(1.0f + __expf(-g)) * u * cws[mbase + r];
        hs[(mbase + r) * HPITCH + n0 + n] = (__bf16)h;
        g = ag1[r]; u = au1[r];
        h = g * __builtin_amdgcn_rcpf(1.0f + __expf(-g)) * u * cws[16 + mbase + r];
        hs[(16 + mbase + r) * HPITCH + n0 + n] = (__bf16)h;
      }
    }
    __syncthreads();

    // ---- Stage 2: out += hidden @ Wd^T (accumulate in registers) -----------
    for (int nt = 0; nt < 8; ++nt) {
      int n0 = wave * 128 + nt * 16;
      for (int k0 = 0; k0 < IDIM; k0 += 32) {
        v16bf a0 = load_a_frag(hs, HPITCH, k0, lane);
        v16bf a1 = load_a_frag(hs + 16 * HPITCH, HPITCH, k0, lane);
        v16bf b  = load_b_frag(wd, IDIM, n0, k0, lane);
        oacc0[nt] = __builtin_amdgcn_wmma_f32_16x16x32_bf16(
            false, a0, false, b, (short)0, oacc0[nt], false, false);
        oacc1[nt] = __builtin_amdgcn_wmma_f32_16x16x32_bf16(
            false, a1, false, b, (short)0, oacc1[nt], false, false);
      }
    }
  }

  // Write output tile (fp32).
  {
    int n = lane & 15, mbase = (lane >> 4) * 8;
#pragma unroll
    for (int nt = 0; nt < 8; ++nt) {
      int n0 = wave * 128 + nt * 16;
#pragma unroll
      for (int r = 0; r < 8; ++r) {
        out[(size_t)(token0 + mbase + r) * HDIM + n0 + n] = oacc0[nt][r];
        out[(size_t)(token0 + 16 + mbase + r) * HDIM + n0 + n] = oacc1[nt][r];
      }
    }
  }
}

// ---------------------------------------------------------------------------
extern "C" void kernel_launch(void* const* d_in, const int* in_sizes, int n_in,
                              void* d_out, int out_size, void* d_ws, size_t ws_size,
                              hipStream_t stream) {
  (void)in_sizes; (void)n_in; (void)out_size; (void)ws_size;

  const float* x    = (const float*)d_in[0];
  const float* Wg_s = (const float*)d_in[1];
  const float* Wu_s = (const float*)d_in[2];
  const float* Wd_s = (const float*)d_in[3];
  const float* Wg   = (const float*)d_in[4];
  const float* Wu   = (const float*)d_in[5];
  const float* Wd   = (const float*)d_in[6];
  const float* Wr   = (const float*)d_in[7];
  const float* rb   = (const float*)d_in[8];
  float* out = (float*)d_out;

  // Workspace carve-up (~147 MB): bf16 copies of x + all weights, plus cw.
  char* w = (char*)d_ws;
  __bf16* xb  = (__bf16*)w; w += (size_t)N_TOK * HDIM * 2;
  __bf16* wgs = (__bf16*)w; w += (size_t)IDIM * HDIM * 2;
  __bf16* wus = (__bf16*)w; w += (size_t)IDIM * HDIM * 2;
  __bf16* wds = (__bf16*)w; w += (size_t)HDIM * IDIM * 2;
  __bf16* wgb = (__bf16*)w; w += (size_t)NEXP * IDIM * HDIM * 2;
  __bf16* wub = (__bf16*)w; w += (size_t)NEXP * IDIM * HDIM * 2;
  __bf16* wdb = (__bf16*)w; w += (size_t)NEXP * HDIM * IDIM * 2;
  float*  cw  = (float*)w;  w += (size_t)N_TOK * NEXP * 4;

  dim3 blk(256);
  cvt_kernel<<<2048, blk, 0, stream>>>(x, xb, (long long)N_TOK * HDIM);
  cvt_kernel<<<512,  blk, 0, stream>>>(Wg_s, wgs, (long long)IDIM * HDIM);
  cvt_kernel<<<512,  blk, 0, stream>>>(Wu_s, wus, (long long)IDIM * HDIM);
  cvt_kernel<<<512,  blk, 0, stream>>>(Wd_s, wds, (long long)HDIM * IDIM);
  cvt_kernel<<<4096, blk, 0, stream>>>(Wg, wgb, (long long)NEXP * IDIM * HDIM);
  cvt_kernel<<<4096, blk, 0, stream>>>(Wu, wub, (long long)NEXP * IDIM * HDIM);
  cvt_kernel<<<4096, blk, 0, stream>>>(Wd, wdb, (long long)NEXP * HDIM * IDIM);

  router_kernel<<<N_TOK / 8, blk, 0, stream>>>(x, Wr, rb, cw);

  (void)hipFuncSetAttribute((const void*)moe_kernel,
                            hipFuncAttributeMaxDynamicSharedMemorySize, MOE_SMEM);
  moe_kernel<<<N_TOK / MTILE, blk, MOE_SMEM, stream>>>(xb, wgs, wus, wds, wgb,
                                                       wub, wdb, cw, out);
}
